// Top_Down_With_Pair_Rf_73796128080257
// MI455X (gfx1250) — matrix operand order repaired and checked by reading, not compiled, not run on hardware
//
#include <hip/hip_runtime.h>
#include <hip/hip_bf16.h>
#include <math.h>

// ---------------------------------------------------------------------------
// Types for CDNA5 WMMA (gfx1250, wave32)
// ---------------------------------------------------------------------------
typedef __attribute__((ext_vector_type(16))) __bf16        v16bf;
typedef __attribute__((ext_vector_type(8)))  float         v8f;
typedef __attribute__((ext_vector_type(4)))  float         f32x4;
typedef __attribute__((ext_vector_type(4)))  unsigned int  u32x4;

union Frag { v16bf v; u32x4 q[2]; };

__device__ __forceinline__ unsigned short f2bf(float f) {
    unsigned int u = __float_as_uint(f);
    u += 0x7fffu + ((u >> 16) & 1u);   // round-to-nearest-even
    return (unsigned short)(u >> 16);
}

__constant__ int cREQ[6][5] = {
    {1,2,3,4,5},{0,2,3,4,5},{0,1,3,4,5},{0,1,2,4,5},{0,1,2,3,5},{0,1,2,3,4}
};

#define BDIM 1024
#define NB_B 128
#define NB_R 6
#define NB_S 49
#define NB_C 2048

// ---------------------------------------------------------------------------
// Generic bf16-WMMA GEMM:  C = act(A(MxK,f32) * B(KxN,f32) + bias)
// Block tile 128x128x32, 256 threads = 8 waves (2x4), wave tile 64x32.
// ---------------------------------------------------------------------------
__launch_bounds__(256)
__global__ void gemm_bf16_wmma(const float* __restrict__ A,
                               const float* __restrict__ B,
                               const float* __restrict__ bias,
                               float* __restrict__ C,
                               int M, int N, int K, int relu)
{
    __shared__ __attribute__((aligned(16))) unsigned short lA[128 * 40];
    __shared__ __attribute__((aligned(16))) unsigned short lB[128 * 40];

    const int tid  = threadIdx.x;
    const int lane = tid & 31;
    const int wave = tid >> 5;
    const int half = lane >> 4;       // 0 or 1
    const int lm   = lane & 15;       // 0..15
    const int wm   = (wave >> 2) * 64;   // 0 or 64
    const int wn   = (wave & 3) * 32;    // 0,32,64,96
    const int m0   = blockIdx.y * 128;
    const int n0   = blockIdx.x * 128;

    v8f acc[4][2];
    #pragma unroll
    for (int i = 0; i < 4; ++i)
        #pragma unroll
        for (int j = 0; j < 2; ++j)
            #pragma unroll
            for (int r = 0; r < 8; ++r) acc[i][j][r] = 0.0f;

    // staging coordinates
    const int arow = tid >> 1;          // 0..127  (A row within tile)
    const int acg  = (tid & 1) * 16;    // 0 or 16 (A col group)
    const int bk   = tid & 31;          // 0..31   (B k within tile)
    const int bng  = (tid >> 5) * 16;   // 0..112  (B col group)

    for (int k0 = 0; k0 < K; k0 += 32) {
        if (k0) __syncthreads();

        // ---- stage A tile: f32 -> bf16 into lA[row][k] (row-major, pad 8) ----
        {
            unsigned short tmp[16];
            const int gRow = m0 + arow;
            const int gk   = k0 + acg;
            if (gRow < M && (gk + 16) <= K && ((K & 3) == 0)) {
                const f32x4* src = (const f32x4*)(A + (size_t)gRow * K + gk);
                #pragma unroll
                for (int j = 0; j < 4; ++j) {
                    f32x4 v = src[j];
                    tmp[j*4+0] = f2bf(v.x); tmp[j*4+1] = f2bf(v.y);
                    tmp[j*4+2] = f2bf(v.z); tmp[j*4+3] = f2bf(v.w);
                }
            } else {
                #pragma unroll
                for (int j = 0; j < 16; ++j) {
                    int kk = gk + j;
                    float v = (gRow < M && kk < K) ? A[(size_t)gRow * K + kk] : 0.0f;
                    tmp[j] = f2bf(v);
                }
            }
            #pragma unroll
            for (int j = 0; j < 16; ++j) lA[arow * 40 + acg + j] = tmp[j];
        }

        // ---- stage B tile transposed: f32 -> bf16 into lB[n][k] ----
        {
            unsigned short tmp[16];
            const int gk      = k0 + bk;
            const int colBase = n0 + bng;
            if (gk < K && (colBase + 16) <= N && ((N & 3) == 0)) {
                const f32x4* src = (const f32x4*)(B + (size_t)gk * N + colBase);
                #pragma unroll
                for (int j = 0; j < 4; ++j) {
                    f32x4 v = src[j];
                    tmp[j*4+0] = f2bf(v.x); tmp[j*4+1] = f2bf(v.y);
                    tmp[j*4+2] = f2bf(v.z); tmp[j*4+3] = f2bf(v.w);
                }
            } else {
                #pragma unroll
                for (int j = 0; j < 16; ++j) {
                    int cc = colBase + j;
                    float v = (gk < K && cc < N) ? B[(size_t)gk * N + cc] : 0.0f;
                    tmp[j] = f2bf(v);
                }
            }
            #pragma unroll
            for (int j = 0; j < 16; ++j) lB[(bng + j) * 40 + bk] = tmp[j];
        }
        __syncthreads();

        // ---- load fragments per ISA 16-bit layouts, issue WMMAs ----
        Frag aF[4], bF[2];
        #pragma unroll
        for (int mt = 0; mt < 4; ++mt) {
            int base = (wm + mt * 16 + lm) * 40 + 8 * half;     // K runs: 8h..8h+7, 16+8h..
            aF[mt].q[0] = *(const u32x4*)&lA[base];
            aF[mt].q[1] = *(const u32x4*)&lA[base + 16];
        }
        #pragma unroll
        for (int nt = 0; nt < 2; ++nt) {
            int base = (wn + nt * 16 + lm) * 40 + 8 * half;     // K runs: 8h..8h+7, 16+8h..
            bF[nt].q[0] = *(const u32x4*)&lB[base];
            bF[nt].q[1] = *(const u32x4*)&lB[base + 16];
        }
        #pragma unroll
        for (int mt = 0; mt < 4; ++mt)
            #pragma unroll
            for (int nt = 0; nt < 2; ++nt)
                acc[mt][nt] = __builtin_amdgcn_wmma_f32_16x16x32_bf16(
                    false, aF[mt].v, false, bF[nt].v,
                    (short)0, acc[mt][nt], false, false);
    }

    // ---- epilogue: bias + optional relu, guarded stores ----
    #pragma unroll
    for (int mt = 0; mt < 4; ++mt) {
        #pragma unroll
        for (int nt = 0; nt < 2; ++nt) {
            int colN = n0 + wn + nt * 16 + lm;
            if (colN >= N) continue;
            float bb = bias ? bias[colN] : 0.0f;
            #pragma unroll
            for (int r = 0; r < 8; ++r) {
                int rowM = m0 + wm + mt * 16 + r + 8 * half;
                if (rowM >= M) continue;
                float v = acc[mt][nt][r] + bb;
                if (relu) v = fmaxf(v, 0.0f);
                C[(size_t)rowM * N + colN] = v;
            }
        }
    }
}

// ---------------------------------------------------------------------------
// rv[n] = concat(verb_table[gt_verb[b]], role_table[role_idx[n]])  (n = b*6+r)
// ---------------------------------------------------------------------------
__global__ void build_rv(const int* __restrict__ gt_verb, const int* __restrict__ role_idx,
                         const float* __restrict__ verb_table, const float* __restrict__ role_table,
                         float* __restrict__ rv)
{
    int n = blockIdx.x; int b = n / NB_R;
    int vb = gt_verb[b]; int rl = role_idx[n];
    for (int e = threadIdx.x; e < 600; e += blockDim.x) {
        float val = (e < 300) ? verb_table[(size_t)vb * 300 + e]
                              : role_table[(size_t)rl * 300 + (e - 300)];
        rv[(size_t)n * 600 + e] = val;
    }
}

// vmat[(b*49+s)*2048 + c] = v_org[b][c][s]
__global__ void transpose_v(const float* __restrict__ v_org, float* __restrict__ vmat)
{
    size_t id = (size_t)blockIdx.x * blockDim.x + threadIdx.x;
    const size_t total = (size_t)NB_B * NB_C * NB_S;
    if (id >= total) return;
    int c = (int)(id % NB_C);
    size_t t = id / NB_C;
    int s = (int)(t % NB_S);
    int b = (int)(t / NB_S);
    vmat[id] = v_org[((size_t)b * NB_C + c) * NB_S + s];
}

// ---------------------------------------------------------------------------
// Per-n attention: logits over S, softmax, weighted pool of img -> v_emb
// ---------------------------------------------------------------------------
__launch_bounds__(256)
__global__ void attention_kernel(const float* __restrict__ vp, const float* __restrict__ qp,
                                 const float* __restrict__ att_w, const float* __restrict__ vmat,
                                 float* __restrict__ v_emb)
{
    int n = blockIdx.x; int b = n / NB_R; int tid = threadIdx.x;
    __shared__ float wsh[BDIM];
    __shared__ float red[256];
    __shared__ float lg[64];

    for (int d = tid; d < BDIM; d += 256)
        wsh[d] = qp[(size_t)n * BDIM + d] * att_w[d];
    __syncthreads();

    for (int s = 0; s < NB_S; ++s) {
        const float* vrow = vp + (size_t)(b * NB_S + s) * BDIM;
        float p = 0.0f;
        for (int d = tid; d < BDIM; d += 256) p += vrow[d] * wsh[d];
        red[tid] = p; __syncthreads();
        for (int off = 128; off > 0; off >>= 1) {
            if (tid < off) red[tid] += red[tid + off];
            __syncthreads();
        }
        if (tid == 0) lg[s] = red[0];
        __syncthreads();
    }

    if (tid == 0) {
        float mx = -1e30f;
        for (int s = 0; s < NB_S; ++s) mx = fmaxf(mx, lg[s]);
        float sm = 0.0f;
        for (int s = 0; s < NB_S; ++s) { float e = expf(lg[s] - mx); lg[s] = e; sm += e; }
        float inv = 1.0f / sm;
        for (int s = 0; s < NB_S; ++s) lg[s] *= inv;
    }
    __syncthreads();

    for (int c = tid; c < NB_C; c += 256) {
        float acc = 0.0f;
        for (int s = 0; s < NB_S; ++s)
            acc += lg[s] * vmat[(size_t)(b * NB_S + s) * NB_C + c];
        v_emb[(size_t)n * NB_C + c] = acc;
    }
}

// ---------------------------------------------------------------------------
// all_roles[n] = l2norm(signed_sqrt(q_repr[n] * v_repr[n]))
// (sum of squares of signed_sqrt == sum |x|)
// ---------------------------------------------------------------------------
__launch_bounds__(256)
__global__ void roles_kernel(const float* __restrict__ q_repr, const float* __restrict__ v_repr,
                             float* __restrict__ all_roles)
{
    int n = blockIdx.x; int tid = threadIdx.x;
    __shared__ float red[256];
    float o[4]; float part = 0.0f;
    #pragma unroll
    for (int i = 0; i < 4; ++i) {
        int d = tid + i * 256;
        float x = q_repr[(size_t)n * BDIM + d] * v_repr[(size_t)n * BDIM + d];
        o[i] = x; part += fabsf(x);
    }
    red[tid] = part; __syncthreads();
    for (int off = 128; off > 0; off >>= 1) {
        if (tid < off) red[tid] += red[tid + off];
        __syncthreads();
    }
    float inv = 1.0f / fmaxf(sqrtf(red[0]), 1e-12f);
    #pragma unroll
    for (int i = 0; i < 4; ++i) {
        int d = tid + i * 256;
        float ss = copysignf(sqrtf(fabsf(o[i])), o[i]);
        all_roles[(size_t)n * BDIM + d] = ss * inv;
    }
}

// ---------------------------------------------------------------------------
// Pair message passing (factorized): context = sum_{k1,k2} relu(X1[nb2]+X2[nb1]+X3[n]+b)
// then gate/update.
// ---------------------------------------------------------------------------
__launch_bounds__(256)
__global__ void pair_kernel(const float* __restrict__ X1, const float* __restrict__ X2,
                            const float* __restrict__ X3, const float* __restrict__ pair_b,
                            const float* __restrict__ all_roles, float* __restrict__ updated)
{
    int n = blockIdx.x; int b = n / NB_R; int r = n % NB_R; int tid = threadIdx.x;
    __shared__ float red[256];
    int nb[5];
    #pragma unroll
    for (int j = 0; j < 5; ++j) nb[j] = b * NB_R + cREQ[r][j];

    float ctx[4], ar[4];
    float part = 0.0f;
    #pragma unroll
    for (int i = 0; i < 4; ++i) {
        int d = tid + i * 256;
        float x3 = X3[(size_t)n * BDIM + d] + pair_b[d];
        float a1[5], a2[5];
        #pragma unroll
        for (int j = 0; j < 5; ++j) {
            a1[j] = X1[(size_t)nb[j] * BDIM + d];
            a2[j] = X2[(size_t)nb[j] * BDIM + d];
        }
        float s = 0.0f;
        #pragma unroll
        for (int k2 = 0; k2 < 5; ++k2)
            #pragma unroll
            for (int k1 = 0; k1 < 5; ++k1) {
                float t = a1[k2] + a2[k1] + x3;
                s += fmaxf(t, 0.0f);
            }
        ctx[i] = s;
        ar[i]  = all_roles[(size_t)n * BDIM + d];
        part  += fabsf(ctx[i] * ar[i]);
    }
    red[tid] = part; __syncthreads();
    for (int off = 128; off > 0; off >>= 1) {
        if (tid < off) red[tid] += red[tid + off];
        __syncthreads();
    }
    float inv = 1.0f / fmaxf(sqrtf(red[0]), 1e-12f);
    #pragma unroll
    for (int i = 0; i < 4; ++i) {
        int d = tid + i * 256;
        float joint = ctx[i] * ar[i];
        float z = copysignf(sqrtf(fabsf(joint)), joint) * inv;
        float g = 1.0f / (1.0f + expf(-z));
        updated[(size_t)n * BDIM + d] = g * ar[i] + (1.0f - g) * ctx[i];
    }
}

// ---------------------------------------------------------------------------
extern "C" void kernel_launch(void* const* d_in, const int* in_sizes, int n_in,
                              void* d_out, int out_size, void* d_ws, size_t ws_size,
                              hipStream_t stream)
{
    (void)in_sizes; (void)n_in; (void)out_size; (void)ws_size;

    const float* v_org      = (const float*)d_in[0];
    const int*   gt_verb    = (const int*)  d_in[1];
    const int*   role_idx   = (const int*)  d_in[2];
    const float* verb_table = (const float*)d_in[3];
    const float* role_table = (const float*)d_in[4];
    const float* qc_W       = (const float*)d_in[5];
    const float* qc_b       = (const float*)d_in[6];
    const float* attv_W     = (const float*)d_in[7];
    const float* attq_W     = (const float*)d_in[8];
    const float* att_w      = (const float*)d_in[9];
    const float* vnet_W     = (const float*)d_in[10];
    const float* vnet_b     = (const float*)d_in[11];
    const float* qnet_W     = (const float*)d_in[12];
    const float* qnet_b     = (const float*)d_in[13];
    const float* pair_W     = (const float*)d_in[14];
    const float* pair_b     = (const float*)d_in[15];
    const float* cls_W      = (const float*)d_in[16];
    const float* cls_b      = (const float*)d_in[17];
    float*       out        = (float*)d_out;

    const int N  = NB_B * NB_R;        // 768
    const int BS = NB_B * NB_S;        // 6272

    float* w = (float*)d_ws;
    size_t off = 0;
    float* rv        = w + off; off += (size_t)N  * 600;
    float* q_emb     = w + off; off += (size_t)N  * 1024;
    float* vmat      = w + off; off += (size_t)BS * 2048;
    float* vp        = w + off; off += (size_t)BS * 1024;
    float* qp        = w + off; off += (size_t)N  * 1024;
    float* v_emb     = w + off; off += (size_t)N  * 2048;
    float* v_repr    = w + off; off += (size_t)N  * 1024;
    float* q_repr    = w + off; off += (size_t)N  * 1024;
    float* all_roles = w + off; off += (size_t)N  * 1024;
    float* X1        = w + off; off += (size_t)N  * 1024;
    float* X2        = w + off; off += (size_t)N  * 1024;
    float* X3        = w + off; off += (size_t)N  * 1024;
    float* updated   = w + off; off += (size_t)N  * 1024;

    auto gemm = [&](const float* A, const float* B, const float* bias, float* C,
                    int M, int Nn, int K, int relu) {
        dim3 grid((Nn + 127) / 128, (M + 127) / 128);
        gemm_bf16_wmma<<<grid, 256, 0, stream>>>(A, B, bias, C, M, Nn, K, relu);
    };

    build_rv<<<N, 256, 0, stream>>>(gt_verb, role_idx, verb_table, role_table, rv);

    {
        size_t total = (size_t)NB_B * NB_C * NB_S;
        transpose_v<<<(unsigned)((total + 255) / 256), 256, 0, stream>>>(v_org, vmat);
    }

    gemm(rv,   qc_W,   qc_b,    q_emb, N,  1024, 600,  1);   // q_emb = relu(rv @ qc_W + b)
    gemm(vmat, attv_W, nullptr, vp,    BS, 1024, 2048, 1);   // vp    = relu(v @ attv_W)
    gemm(q_emb, attq_W, nullptr, qp,   N,  1024, 1024, 1);   // qp    = relu(q_emb @ attq_W)

    attention_kernel<<<N, 256, 0, stream>>>(vp, qp, att_w, vmat, v_emb);

    gemm(v_emb, vnet_W, vnet_b, v_repr, N, 1024, 2048, 1);   // v_repr
    gemm(q_emb, qnet_W, qnet_b, q_repr, N, 1024, 1024, 1);   // q_repr

    roles_kernel<<<N, 256, 0, stream>>>(q_repr, v_repr, all_roles);

    // pair_W (3072x1024) splits row-wise into W1, W2, W3
    gemm(all_roles, pair_W,                        nullptr, X1, N, 1024, 1024, 0);
    gemm(all_roles, pair_W + (size_t)1024 * 1024,  nullptr, X2, N, 1024, 1024, 0);
    gemm(all_roles, pair_W + (size_t)2048 * 1024,  nullptr, X3, N, 1024, 1024, 0);

    pair_kernel<<<N, 256, 0, stream>>>(X1, X2, X3, pair_b, all_roles, updated);

    gemm(updated, cls_W, cls_b, out, N, 2001, 1024, 0);      // final logits
}
